// BinaryConv2D_38551626449709
// MI455X (gfx1250) — matrix operand order repaired and checked by reading, not compiled, not run on hardware
//
#include <hip/hip_runtime.h>

typedef _Float16 v16h __attribute__((ext_vector_type(16)));
typedef float    v8f  __attribute__((ext_vector_type(8)));
typedef int      v4i  __attribute__((ext_vector_type(4)));
typedef __attribute__((address_space(1))) v4i g_v4i;   // global (AS1) int4
typedef __attribute__((address_space(3))) v4i l_v4i;   // LDS    (AS3) int4

// Problem constants
#define NIMG 32
#define HIMG 56
#define WIMG 56
#define CIN  256
#define COUT 256
#define HWPLANE (HIMG * WIMG)          // 3136
#define MTOT (NIMG * HWPLANE)          // 100352
#define BM 128
#define BN 128
#define KTILE 32
#define NSTAGES 72                     // 9 taps * 8 ci-chunks
#define ASTRIDE 36                     // f32 LDS row stride (32 + 4 pad), 144B (16B aligned)
// prepacked B: [stage(72)][ntile(16)][lane(32)][elem(16)] f16
#define B_TILE_HALFS 512
#define B_TOTAL_HALFS (NSTAGES * 16 * B_TILE_HALFS)  // 589824

#if defined(__gfx1250__) && \
    __has_builtin(__builtin_amdgcn_global_load_async_to_lds_b128) && \
    __has_builtin(__builtin_amdgcn_s_wait_asynccnt)
#define HAVE_ASYNC 1
#else
#define HAVE_ASYNC 0
#endif

// ---------------------------------------------------------------------------
// Prepack: binarize fp32 HWIO weights -> f16 B fragments (contiguous k order, same as A).
// B tile (K=32 x N=16): lanes 0-15 hold K=0..15 (N=lane), lanes 16-31 hold
// K=16..31 (N=lane-16); 16 contiguous halves per lane.
// ---------------------------------------------------------------------------
__global__ __launch_bounds__(256) void binprepack_kernel(
    const float* __restrict__ w, _Float16* __restrict__ wsB) {
  int idx = blockIdx.x * 256 + threadIdx.x;           // 0 .. 589823 (== 9<<16)
  int i    = idx & 15;
  int lane = (idx >> 4) & 31;
  int nt   = (idx >> 9) & 15;
  int kc   = (idx >> 13) & 7;
  int tap  = idx >> 16;                               // 0..8
  int ci = kc * 32 + (lane >> 4) * 16 + i;
  int co = nt * 16 + (lane & 15);
  float v = w[((size_t)(tap * CIN + ci)) * COUT + co];
  float s = (v > 0.0f) ? 1.0f : ((v < 0.0f) ? -1.0f : 0.0f);
  wsB[idx] = (_Float16)s;
}

// ---------------------------------------------------------------------------
// Implicit-GEMM binary conv, V_WMMA_F32_16X16X32_F16, async-LDS double buffer.
// Block: 128M x 128N, 256 threads = 8 waves (4 m-waves x 2 n-waves),
// each wave computes 2x4 16x16 tiles. A staged as f32 via async copy,
// converted to f16 during fragment assembly (co-executes with WMMA).
// ---------------------------------------------------------------------------
__global__ __launch_bounds__(256) void binconv_wmma_kernel(
    const float* __restrict__ x, const _Float16* __restrict__ wsB,
    float* __restrict__ out) {
  __shared__ float sAf[2][BM][ASTRIDE];               // 36,864 B double buffer
  __shared__ float sDump[256 * 4];                    // async dump slots (4 KB)

  const int tid  = threadIdx.x;
  const int lane = tid & 31;
  const int wave = tid >> 5;
  const int wm   = wave >> 1;                         // 0..3
  const int wn   = wave & 1;                          // 0..1

  // A staging: 2 threads per position, 16 f32 each
  const int posA  = tid >> 1;
  const int halfA = tid & 1;
  const int p     = blockIdx.x * BM + posA;
  const int nImg  = p / HWPLANE;
  const int hw    = p % HWPLANE;
  const int oh    = hw / WIMG;
  const int ow    = hw % WIMG;
  const float* xbase = x + (size_t)(nImg * HWPLANE + oh * WIMG + ow) * CIN;

  const int rowSel = lane & 15;
  const int khalf  = lane >> 4;
  const int ntBase = blockIdx.y * 8 + wn * 4;

  v8f acc[2][4] = {};

  // ---- stage issue: async copy (or sync fallback) into buf[s&1] ----
  auto issue = [&](int s) {
    const int tap = s >> 3;
    const int kc  = s & 7;
    const int kh = tap / 3, kw = tap - kh * 3;
    const int ih = oh + kh - 1, iw = ow + kw - 1;
    const bool valid = ((unsigned)ih < (unsigned)HIMG) &&
                       ((unsigned)iw < (unsigned)WIMG);
    float* realDst = &sAf[s & 1][posA][halfA * 16];
    const float* realSrc =
        xbase + ((kh - 1) * WIMG + (kw - 1)) * CIN + kc * KTILE + halfA * 16;
#if HAVE_ASYNC
    if (!valid) {                     // zeros via normal DS stores
      float4 z; z.x = z.y = z.z = z.w = 0.0f;
      float4* d4 = (float4*)realDst;
      d4[0] = z; d4[1] = z; d4[2] = z; d4[3] = z;
    }
    // Every lane issues 4 asyncs so every wave's batch is exactly 4
    // (wave-uniform ASYNCcnt => s_wait_asynccnt(4) == "previous stage done").
    // Invalid lanes copy from a safe address into a private dump slot.
    float* dumpSlot = &sDump[tid * 4];
#pragma unroll
    for (int j = 0; j < 4; ++j) {
      const float* gs = valid ? (realSrc + 4 * j) : xbase;
      float*       ld = valid ? (realDst + 4 * j) : dumpSlot;
      __builtin_amdgcn_global_load_async_to_lds_b128(
          (g_v4i*)gs, (l_v4i*)ld, 0, 0);
    }
#else
    float4 v0, v1, v2, v3;
    if (valid) {
      const float4* s4 = (const float4*)realSrc;
      v0 = s4[0]; v1 = s4[1]; v2 = s4[2]; v3 = s4[3];
    } else {
      v0.x = v0.y = v0.z = v0.w = 0.0f; v1 = v0; v2 = v0; v3 = v0;
    }
    float4* d4 = (float4*)realDst;
    d4[0] = v0; d4[1] = v1; d4[2] = v2; d4[3] = v3;
#endif
  };

  // ---- compute: A fragments from LDS (f32 -> f16), B from global, WMMA ----
  auto compute = [&](int s) {
    v16h a[2];
#pragma unroll
    for (int ta = 0; ta < 2; ++ta) {
      const float* rb = &sAf[s & 1][wm * 32 + ta * 16 + rowSel][0];
      float4 c0 = *(const float4*)(rb + khalf * 16);
      float4 c1 = *(const float4*)(rb + khalf * 16 + 4);
      float4 c2 = *(const float4*)(rb + khalf * 16 + 8);
      float4 c3 = *(const float4*)(rb + khalf * 16 + 12);
      v16h av;
      av[0]  = (_Float16)c0.x; av[1]  = (_Float16)c0.y;
      av[2]  = (_Float16)c0.z; av[3]  = (_Float16)c0.w;
      av[4]  = (_Float16)c1.x; av[5]  = (_Float16)c1.y;
      av[6]  = (_Float16)c1.z; av[7]  = (_Float16)c1.w;
      av[8]  = (_Float16)c2.x; av[9]  = (_Float16)c2.y;
      av[10] = (_Float16)c2.z; av[11] = (_Float16)c2.w;
      av[12] = (_Float16)c3.x; av[13] = (_Float16)c3.y;
      av[14] = (_Float16)c3.z; av[15] = (_Float16)c3.w;
      a[ta] = av;
    }
    const _Float16* wst = wsB + (size_t)s * (16 * B_TILE_HALFS) + lane * 16;
    if (s + 1 < NSTAGES)  // warm L0/WGP$ for next stage's B fragments
      __builtin_prefetch(wst + 16 * B_TILE_HALFS, 0, 0);
#pragma unroll
    for (int tb = 0; tb < 4; ++tb) {
      v16h b = *(const v16h*)(wst + (size_t)(ntBase + tb) * B_TILE_HALFS);
#pragma unroll
      for (int ta = 0; ta < 2; ++ta) {
        acc[ta][tb] = __builtin_amdgcn_wmma_f32_16x16x32_f16(
            false, a[ta], false, b, (short)0, acc[ta][tb], false, false);
      }
    }
  };

  // ---- 2-deep software pipeline over 72 K-stages ----
  issue(0);
  issue(1);
#pragma unroll 2   // make s&1 a compile-time constant per copy (static LDS offsets)
  for (int s = 0; s < NSTAGES - 1; ++s) {
#if HAVE_ASYNC
    __builtin_amdgcn_s_wait_asynccnt(4);   // batch s landed (batch s+1 in flight)
#endif
    __syncthreads();
    compute(s);
    __syncthreads();
    issue(s + 2 < NSTAGES ? s + 2 : s);    // s+2==NSTAGES only at s==70
  }
#if HAVE_ASYNC
  __builtin_amdgcn_s_wait_asynccnt(0);
#endif
  __syncthreads();
  compute(NSTAGES - 1);

  // ---- epilogue: C layout — VGPR r holds M=r (lanes 0-15) / M=r+8 ----
  // Non-temporal: output is write-once, keep it from evicting L2-resident
  // input/weight streams.
  const int mBase = blockIdx.x * BM + wm * 32;
  const int cBase = blockIdx.y * BN + wn * 64;
#pragma unroll
  for (int ta = 0; ta < 2; ++ta)
#pragma unroll
    for (int tb = 0; tb < 4; ++tb)
#pragma unroll
      for (int r = 0; r < 8; ++r) {
        const int m = mBase + ta * 16 + r + khalf * 8;
        const int c = cBase + tb * 16 + rowSel;
        __builtin_nontemporal_store(acc[ta][tb][r], &out[(size_t)m * COUT + c]);
      }
}

// ---------------------------------------------------------------------------
// Fallback (only if workspace too small for prepacked weights)
// ---------------------------------------------------------------------------
__global__ void binconv_naive_kernel(const float* __restrict__ x,
                                     const float* __restrict__ w,
                                     float* __restrict__ out, int total) {
  int idx = blockIdx.x * blockDim.x + threadIdx.x;
  if (idx >= total) return;
  int co = idx & (COUT - 1);
  int p  = idx >> 8;
  int n  = p / HWPLANE;
  int hw = p % HWPLANE;
  int oh = hw / WIMG, ow = hw % WIMG;
  float acc = 0.0f;
  for (int kh = 0; kh < 3; ++kh) {
    int ih = oh + kh - 1;
    if ((unsigned)ih >= (unsigned)HIMG) continue;
    for (int kw = 0; kw < 3; ++kw) {
      int iw = ow + kw - 1;
      if ((unsigned)iw >= (unsigned)WIMG) continue;
      const float* xs = x + (size_t)(n * HWPLANE + ih * WIMG + iw) * CIN;
      const float* wp = w + (size_t)((kh * 3 + kw) * CIN) * COUT + co;
      for (int ci = 0; ci < CIN; ++ci) {
        float wv = wp[(size_t)ci * COUT];
        float s = (wv > 0.0f) ? 1.0f : ((wv < 0.0f) ? -1.0f : 0.0f);
        acc += xs[ci] * s;
      }
    }
  }
  out[idx] = acc;
}

extern "C" void kernel_launch(void* const* d_in, const int* in_sizes, int n_in,
                              void* d_out, int out_size, void* d_ws, size_t ws_size,
                              hipStream_t stream) {
  const float* x = (const float*)d_in[0];   // (32,56,56,256) f32 NHWC
  const float* w = (const float*)d_in[1];   // (3,3,256,256)  f32 HWIO
  float* out = (float*)d_out;               // (32,56,56,256) f32

  const size_t needB = (size_t)B_TOTAL_HALFS * sizeof(_Float16);  // ~1.13 MB
  if (ws_size >= needB) {
    _Float16* wsB = (_Float16*)d_ws;
    binprepack_kernel<<<B_TOTAL_HALFS / 256, 256, 0, stream>>>(w, wsB);
    dim3 grid(MTOT / BM, COUT / BN);        // (784, 2)
    binconv_wmma_kernel<<<grid, 256, 0, stream>>>(x, wsB, out);
  } else {
    int total = MTOT * COUT;
    binconv_naive_kernel<<<(total + 255) / 256, 256, 0, stream>>>(x, w, out, total);
  }
}
